// SelfAttention_36885179138225
// MI455X (gfx1250) — matrix operand order repaired and checked by reading, not compiled, not run on hardware
//
#include <hip/hip_runtime.h>
#include <stdint.h>

#define SEQ_LEN 4096
#define EMB     256
#define BATCH   4
#define TSTEP   32
#define NTS     (SEQ_LEN / TSTEP)
#define WAVES   8
#define ROWS_PER_WG (WAVES * 16)   // 128 query rows per workgroup

typedef __attribute__((ext_vector_type(16))) __bf16    v16bf;
typedef __attribute__((ext_vector_type(8)))  float     v8f;
typedef __attribute__((ext_vector_type(4)))  uint32_t  u32x4;
typedef __attribute__((ext_vector_type(8)))  int       i32x8;
typedef __attribute__((ext_vector_type(4)))  int       i32x4;

union BF16Tile {
    v16bf    v;      // operand for v_wmma_f32_16x16x32_bf16
    uint32_t u[8];
    uint16_t h[16];
    u32x4    q[2];   // two 128-bit halves for b128 loads/stores
};

__device__ inline uint16_t f32_to_bf16(float f) {
    uint32_t u = __float_as_uint(f);
    u += 0x7FFFu + ((u >> 16) & 1u);   // round-to-nearest-even
    return (uint16_t)(u >> 16);
}

// ---------------------------------------------------------------------------
// Kernel A: QKV projection.  One wave computes a 16(tokens) x 16(cols) tile of
// gather(emb_table, x) @ W + b via 8x v_wmma_f32_16x16x32_bf16 (K=256).
// Q,K written row-major bf16 [b][s][e]; V written transposed bf16 [b][e][t].
// ---------------------------------------------------------------------------
__global__ __launch_bounds__(32) void qkv_proj_kernel(
    const int* __restrict__ x, const float* __restrict__ emb_table,
    const float* __restrict__ W, const float* __restrict__ bias,
    uint16_t* __restrict__ out, int transposeOut)
{
    const int lane = threadIdx.x & 31;
    const int half = lane >> 4;
    const int l15  = lane & 15;
    const int b    = blockIdx.x / (SEQ_LEN / 16);
    const int m0   = (blockIdx.x % (SEQ_LEN / 16)) * 16;
    const int n0   = blockIdx.y * 16;

    const int tok = x[b * SEQ_LEN + m0 + l15];
    const float* embrow = emb_table + (size_t)tok * EMB;

    v8f acc;
    #pragma unroll
    for (int i = 0; i < 8; ++i) acc[i] = 0.0f;

    for (int c = 0; c < 8; ++c) {            // K loop over E in steps of 32
        const int e0 = 32 * c;
        BF16Tile A, Bm;
        #pragma unroll
        for (int j = 0; j < 8; ++j) {
            const int ka = e0 + half * 16 + 2 * j;
            A.h[2 * j]     = f32_to_bf16(embrow[ka]);
            A.h[2 * j + 1] = f32_to_bf16(embrow[ka + 1]);
            const int kb = e0 + half * 16 + 2 * j;
            Bm.h[2 * j]     = f32_to_bf16(W[(size_t)kb * EMB + n0 + l15]);
            Bm.h[2 * j + 1] = f32_to_bf16(W[(size_t)(kb + 1) * EMB + n0 + l15]);
        }
        acc = __builtin_amdgcn_wmma_f32_16x16x32_bf16(
                  false, A.v, false, Bm.v, (short)0, acc, false, false);
    }

    const float bvv = bias[n0 + l15];

    if (!transposeOut) {
        #pragma unroll
        for (int r = 0; r < 8; ++r) {
            const int s = m0 + r + 8 * half;
            out[(size_t)(b * SEQ_LEN + s) * EMB + n0 + l15] =
                f32_to_bf16(acc[r] + bvv);
        }
    } else {
        BF16Tile o;
        #pragma unroll
        for (int r = 0; r < 8; ++r) o.h[r] = f32_to_bf16(acc[r] + bvv);
        uint16_t* dst = out + ((size_t)(b * EMB + n0 + l15)) * SEQ_LEN
                            + m0 + 8 * half;
        *(u32x4*)dst = o.q[0];
    }
}

// ---------------------------------------------------------------------------
// TDM: build a 2D-tile D# per CDNA5 ISA §8 and issue TENSOR_LOAD_TO_LDS.
// data_size=3 (8-byte units). group0: count=1 | lds_addr | 57b global | type=2.
// group1: dims/strides packed at the documented bit offsets. groups 2/3 = 0.
// This toolchain exposes the 6-arg builtin (extra zero-filled i32x8 + cpol).
// ---------------------------------------------------------------------------
#if __has_builtin(__builtin_amdgcn_tensor_load_to_lds)
#define HAVE_TDM 1
#endif

__device__ inline uint32_t lds_off(const void* p) {
    // low 32 bits of a flat shared-aperture address are the LDS byte offset
    return (uint32_t)(uintptr_t)p;
}

#if HAVE_TDM
__device__ inline void tdm_load_2d(uint32_t ldsOff, const void* gptr,
                                   uint32_t tensor_d0, uint32_t tensor_d1,
                                   uint32_t tile_d0, uint32_t tile_d1,
                                   uint64_t stride0)   // all in 8B units
{
    const uint64_t ga = (uint64_t)(uintptr_t)gptr;
    u32x4 g0;
    g0[0] = 1u;                                   // count=1 (valid user D#)
    g0[1] = ldsOff;                               // lds_addr (bytes)
    g0[2] = (uint32_t)ga;                         // global_addr[31:0]
    g0[3] = (uint32_t)(ga >> 32) | 0x80000000u;   // global_addr[56:32]|type=2
    i32x8 g1;
    g1[0] = (int)(3u << 16);                      // data_size=3 (8B), mask=0
    g1[1] = (int)((tensor_d0 & 0xFFFFu) << 16);   // tensor_dim0[15:0]
    g1[2] = (int)((tensor_d0 >> 16) | ((tensor_d1 & 0xFFFFu) << 16));
    g1[3] = (int)((tensor_d1 >> 16) | (tile_d0 << 16));
    g1[4] = (int)(tile_d1 & 0xFFFFu);             // tile_dim1 | tile_dim2=0
    g1[5] = (int)(uint32_t)stride0;               // tensor_dim0_stride[31:0]
    g1[6] = (int)(uint32_t)(stride0 >> 32);       // stride[47:32]|d1stride=0
    g1[7] = 0;
    i32x4 z4;
    z4[0] = 0; z4[1] = 0; z4[2] = 0; z4[3] = 0;
    i32x8 z8;
    #pragma unroll
    for (int i = 0; i < 8; ++i) z8[i] = 0;
    __builtin_amdgcn_tensor_load_to_lds(g0, g1, z4, z4, z8, 0);
}
#endif

// ---------------------------------------------------------------------------
// Kernel B: fused attention, 8 waves per workgroup sharing K/V tiles in LDS.
// Wave w owns 16 query rows; per t-step (32 keys) the TDM double-buffers
// K tile (32x256 bf16, contiguous rows) and V^T tile (256 rows x 64B) into
// LDS; waves read them as b128 ds loads in WMMA operand layout.
// S^T = K_tile x Q^T keeps exp(P) in the A-operand layout for O += P x V.
// ---------------------------------------------------------------------------
__global__ __launch_bounds__(WAVES * 32) void attn_kernel(
    const uint16_t* __restrict__ Qbf, const uint16_t* __restrict__ Kbf,
    const uint16_t* __restrict__ Vt, float* __restrict__ out)
{
    __shared__ uint16_t ldsK[2][TSTEP * EMB];   // 2 x 16 KB
    __shared__ uint16_t ldsV[2][EMB * TSTEP];   // 2 x 16 KB

    const int tid  = threadIdx.x;
    const int wave = tid >> 5;
    const int lane = tid & 31;
    const int half = lane >> 4;
    const int l15  = lane & 15;
    const int b    = blockIdx.x / (SEQ_LEN / ROWS_PER_WG);
    const int m0   = (blockIdx.x % (SEQ_LEN / ROWS_PER_WG)) * ROWS_PER_WG
                     + wave * 16;

    // ---- Preload this wave's Q as WMMA-B tiles (N = 16 query rows) ----
    BF16Tile QB[8];
    {
        const uint16_t* qrow = Qbf + (size_t)(b * SEQ_LEN + m0 + l15) * EMB;
        #pragma unroll
        for (int c = 0; c < 8; ++c) {
            const uint16_t* p = qrow + 32 * c + 8 * half;
            QB[c].q[0] = *(const u32x4*)(p);
            QB[c].q[1] = *(const u32x4*)(p + 16);
        }
    }

    BF16Tile ones;
    #pragma unroll
    for (int i = 0; i < 16; ++i) ones.h[i] = 0x3F80;  // bf16 1.0

    v8f Oacc[16];
    #pragma unroll
    for (int nt = 0; nt < 16; ++nt)
        #pragma unroll
        for (int i = 0; i < 8; ++i) Oacc[nt][i] = 0.0f;
    v8f denom;
    #pragma unroll
    for (int i = 0; i < 8; ++i) denom[i] = 0.0f;

    const float inv = 0.0625f;  // 1/sqrt(256)
    const uint16_t* Kbase = Kbf + (size_t)b * SEQ_LEN * EMB;
    const uint16_t* Vbase = Vt  + (size_t)b * EMB * SEQ_LEN;

#if HAVE_TDM
    if (wave == 0) {  // prime buffer 0
        tdm_load_2d(lds_off(ldsK[0]), Kbase,
                    EMB / 4, SEQ_LEN, EMB / 4, TSTEP, EMB / 4);
        tdm_load_2d(lds_off(ldsV[0]), Vbase,
                    SEQ_LEN / 4, EMB, TSTEP / 4, EMB, SEQ_LEN / 4);
    }
#endif

    for (int ts = 0; ts < NTS; ++ts) {
        const int cur = ts & 1;
#if HAVE_TDM
        if (wave == 0) {
            if (ts + 1 < NTS) {
                // double-buffer: start next tile, then wait for current pair
                tdm_load_2d(lds_off(ldsK[1 - cur]),
                            Kbase + (size_t)(ts + 1) * TSTEP * EMB,
                            EMB / 4, SEQ_LEN, EMB / 4, TSTEP, EMB / 4);
                tdm_load_2d(lds_off(ldsV[1 - cur]),
                            Vbase + (size_t)(ts + 1) * TSTEP,
                            SEQ_LEN / 4, EMB, TSTEP / 4, EMB, SEQ_LEN / 4);
                __builtin_amdgcn_s_wait_tensorcnt(2);
            } else {
                __builtin_amdgcn_s_wait_tensorcnt(0);
            }
        }
        __syncthreads();
        const uint16_t* kb = ldsK[cur];
        const uint16_t* vb = ldsV[cur];
#else
        // Fallback: cooperative synchronous copy into buffer 0.
        {
            const uint16_t* kg = Kbase + (size_t)ts * TSTEP * EMB;
            #pragma unroll
            for (int i = 0; i < 4; ++i) {  // K tile is one contiguous 16 KB
                const int idx = tid + 256 * i;           // 16B units
                *(u32x4*)(&ldsK[0][idx * 8]) = *(const u32x4*)(kg + idx * 8);
            }
            #pragma unroll
            for (int i = 0; i < 4; ++i) {  // V tile: 256 rows x 64B
                const int idx = tid + 256 * i;
                const int row = idx >> 2, col = idx & 3;
                *(u32x4*)(&ldsV[0][idx * 8]) =
                    *(const u32x4*)(Vbase + (size_t)row * SEQ_LEN
                                    + (size_t)ts * TSTEP + col * 8);
            }
        }
        __syncthreads();
        const uint16_t* kb = ldsK[0];
        const uint16_t* vb = ldsV[0];
#endif

        // ---- scores: S^T(16t x 16s) twice, K-loop over E, from LDS ----
        v8f S0, S1;
        #pragma unroll
        for (int i = 0; i < 8; ++i) { S0[i] = 0.0f; S1[i] = 0.0f; }
        #pragma unroll
        for (int c = 0; c < 8; ++c) {
            BF16Tile KA0, KA1;
            const uint16_t* p0 = kb + l15 * EMB + 32 * c + 8 * half;
            const uint16_t* p1 = kb + (16 + l15) * EMB + 32 * c + 8 * half;
            KA0.q[0] = *(const u32x4*)(p0);
            KA0.q[1] = *(const u32x4*)(p0 + 16);
            KA1.q[0] = *(const u32x4*)(p1);
            KA1.q[1] = *(const u32x4*)(p1 + 16);
            S0 = __builtin_amdgcn_wmma_f32_16x16x32_bf16(
                     false, KA0.v, false, QB[c].v, (short)0, S0, false, false);
            S1 = __builtin_amdgcn_wmma_f32_16x16x32_bf16(
                     false, KA1.v, false, QB[c].v, (short)0, S1, false, false);
        }

        // ---- naive softmax numerator (no max subtraction, like ref) ----
        v8f P0, P1;
        #pragma unroll
        for (int i = 0; i < 8; ++i) {
            P0[i] = __expf(S0[i] * inv);
            P1[i] = __expf(S1[i] * inv);
        }

        // In-lane repack: D-layout of S^T == A-layout of P
        BF16Tile Pa;
        #pragma unroll
        for (int j = 0; j < 4; ++j) {
            Pa.h[2 * j]     = f32_to_bf16(P0[2 * j]);
            Pa.h[2 * j + 1] = f32_to_bf16(P0[2 * j + 1]);
            Pa.h[2 * j + 8] = f32_to_bf16(P1[2 * j]);
            Pa.h[2 * j + 9] = f32_to_bf16(P1[2 * j + 1]);
        }

        // ---- denominator: rowsum(P) via WMMA with all-ones B ----
        denom = __builtin_amdgcn_wmma_f32_16x16x32_bf16(
                    false, Pa.v, false, ones.v, (short)0, denom, false, false);

        // ---- O += P x V over all 16 column tiles of E, V from LDS ----
        #pragma unroll
        for (int nt = 0; nt < 16; ++nt) {
            BF16Tile Vb;
            const uint16_t* pv = vb + (nt * 16 + l15) * TSTEP + 8 * half;
            Vb.q[0] = *(const u32x4*)(pv);
            Vb.q[1] = *(const u32x4*)(pv + 16);
            Oacc[nt] = __builtin_amdgcn_wmma_f32_16x16x32_bf16(
                           false, Pa.v, false, Vb.v, (short)0, Oacc[nt],
                           false, false);
        }
        __syncthreads();   // all waves done reading before buffer reuse
    }

    // ---- finalize: O / denom (VGPR r -> row s = r + 8*half) ----
    v8f rdenom;
    #pragma unroll
    for (int r = 0; r < 8; ++r) rdenom[r] = 1.0f / denom[r];

    float* orow = out + (size_t)(b * SEQ_LEN + m0) * EMB;
    #pragma unroll
    for (int nt = 0; nt < 16; ++nt)
        #pragma unroll
        for (int r = 0; r < 8; ++r)
            orow[(size_t)(r + 8 * half) * EMB + nt * 16 + l15] =
                Oacc[nt][r] * rdenom[r];
}

// ---------------------------------------------------------------------------
extern "C" void kernel_launch(void* const* d_in, const int* in_sizes, int n_in,
                              void* d_out, int out_size, void* d_ws,
                              size_t ws_size, hipStream_t stream)
{
    const int*   x   = (const int*)d_in[0];
    const float* emb = (const float*)d_in[1];
    const float* Wq  = (const float*)d_in[2];
    const float* bq  = (const float*)d_in[3];
    const float* Wk  = (const float*)d_in[4];
    const float* bk  = (const float*)d_in[5];
    const float* Wv  = (const float*)d_in[6];
    const float* bv  = (const float*)d_in[7];
    float* out = (float*)d_out;

    const size_t elems = (size_t)BATCH * SEQ_LEN * EMB;
    uint16_t* Qbf = (uint16_t*)d_ws;          //  8 MB bf16 [b][s][e]
    uint16_t* Kbf = Qbf + elems;              //  8 MB bf16 [b][t][e]
    uint16_t* Vt  = Kbf + elems;              //  8 MB bf16 [b][e][t]

    dim3 gridA(BATCH * SEQ_LEN / 16, EMB / 16, 1);
    qkv_proj_kernel<<<gridA, 32, 0, stream>>>(x, emb, Wq, bq, Qbf, 0);
    qkv_proj_kernel<<<gridA, 32, 0, stream>>>(x, emb, Wk, bk, Kbf, 0);
    qkv_proj_kernel<<<gridA, 32, 0, stream>>>(x, emb, Wv, bv, Vt, 1);

    dim3 gridB(BATCH * SEQ_LEN / ROWS_PER_WG, 1, 1);
    attn_kernel<<<gridB, WAVES * 32, 0, stream>>>(Qbf, Kbf, Vt, out);
}
